// LatentReverb_32598801776740
// MI455X (gfx1250) — matrix operand order repaired and checked by reading, not compiled, not run on hardware
//
#include <hip/hip_runtime.h>
#include <hip/hip_bf16.h>
#include <stdint.h>

typedef __attribute__((ext_vector_type(16))) _Float16 v16h;
typedef __attribute__((ext_vector_type(8)))  float    v8f;
typedef uint32_t u32x4 __attribute__((ext_vector_type(4)));
typedef uint32_t u32x8 __attribute__((ext_vector_type(8)));

#define Bn 4
#define Cn 4
#define Hn 64
#define Wn 64
#define Sn 4096   // H*W

__device__ __forceinline__ float sigm(float x) { return 1.0f / (1.0f + __expf(-x)); }

// fade(n=64, fw=4): linspace(0.6,1.0,4) head, linspace(1.0,0.6,4) tail
__device__ __forceinline__ float fade4(int i) {
    if (i < 4)   return 0.6f + (0.4f / 3.0f) * (float)i;
    if (i >= 60) return 1.0f - (0.4f / 3.0f) * (float)(i - 60);
    return 1.0f;
}

// ---------------------------------------------------------------------------
// spatial_features = conv3x3(x, spatial_w[8,4,3,3]) + spatial_b   (zero pad)
// ---------------------------------------------------------------------------
__global__ void lr_spatial(const float* __restrict__ x,
                           const float* __restrict__ spW,
                           const float* __restrict__ spB,
                           float* __restrict__ sp) {
    int idx = blockIdx.x * 256 + threadIdx.x;     // [b*8+oc][p]
    int p  = idx & 4095;
    int bc = idx >> 12;
    int oc = bc & 7;
    int b  = bc >> 3;
    int y = p >> 6, xx = p & 63;
    float acc = spB[oc];
    for (int dy = -1; dy <= 1; ++dy) {
        int yy = y + dy; if ((unsigned)yy >= 64u) continue;
        for (int dx = -1; dx <= 1; ++dx) {
            int x2 = xx + dx; if ((unsigned)x2 >= 64u) continue;
            #pragma unroll
            for (int ic = 0; ic < 4; ++ic)
                acc += spW[((oc*4 + ic)*3 + (dy+1))*3 + (dx+1)]
                     * x[((b*Cn + ic) << 12) + (yy << 6) + x2];
        }
    }
    sp[idx] = acc;
}

// ---------------------------------------------------------------------------
// Full NREF=16 reflection loop for one batch element, entirely in LDS.
// block = 1024 threads, grid = 4.  Dynamic LDS: fb|r|wet (3*16384) + 176 wts.
// Initial x tile (64 KB contiguous) staged via the Tensor Data Mover:
// one wave issues tensor_load_to_lds with a 1-row D#, waits TENSORcnt.
// ---------------------------------------------------------------------------
__global__ void lr_reverb(const float* __restrict__ x,
                          const float* __restrict__ rw,
                          const float* __restrict__ rd,
                          const float* __restrict__ fbW, const float* __restrict__ fbB,
                          const float* __restrict__ d1W, const float* __restrict__ d1B,
                          const float* __restrict__ d2W, const float* __restrict__ d2B,
                          float* __restrict__ wet_out) {
    extern __shared__ float sm[];
    float* fb  = sm;                 // [4][4096]
    float* r   = sm + Cn * Sn;       // [4][4096]
    float* wet = sm + 2 * Cn * Sn;   // [4][4096]
    float* cw  = sm + 3 * Cn * Sn;   // 170 floats of weights
    const int b = blockIdx.x;
    const int t = threadIdx.x;

    // --- TDM: DMA x[b] (16384 floats) into the 'r' LDS region (wave 0 only) ---
    if (t < 32) {
        uint64_t gaddr = (uint64_t)(uintptr_t)(x + ((size_t)b << 14));
        uint32_t ldsd  = (uint32_t)(uintptr_t)r;   // flat LDS addr[31:0] = LDS offset
        u32x4 g0;
        g0[0] = 1u;                                   // count=1, user mode
        g0[1] = ldsd;                                 // lds_addr
        g0[2] = (uint32_t)gaddr;                      // global_addr[31:0]
        g0[3] = (uint32_t)(gaddr >> 32) | (2u << 30); // global_addr[56:32] | type=2
        u32x8 g1;
        g1[0] = 0x00020000u;                          // data_size = 4B, no multicast
        g1[1] = (16384u & 0xffffu) << 16;             // tensor_dim0[15:0]
        g1[2] = (16384u >> 16) | (1u << 16);          // tensor_dim0[31:16] | tensor_dim1[15:0]
        g1[3] = (16384u << 16);                       // tile_dim0 = 16384
        g1[4] = 1u;                                   // tile_dim1 = 1
        g1[5] = 16384u;                               // tensor_dim0_stride lo32
        g1[6] = 0u;
        g1[7] = 0u;
        asm volatile("tensor_load_to_lds %0, %1" :: "s"(g0), "s"(g1) : "memory");
        __builtin_amdgcn_s_wait_tensorcnt(0);
    }

    if (t < 144) cw[t]       = fbW[t];   // feedback_w [4][4][3][3]
    if (t < 4)   cw[144 + t] = fbB[t];
    if (t < 8)   cw[148 + t] = d1W[t];   // [2][4]
    if (t < 2)   cw[156 + t] = d1B[t];
    if (t < 8)   cw[158 + t] = d2W[t];   // [4][2]
    if (t < 4)   cw[166 + t] = d2B[t];
    __syncthreads();

    for (int kk = 0; kk < 4; ++kk) {
        int p = kk * 1024 + t;
        #pragma unroll
        for (int c = 0; c < Cn; ++c) {
            fb[c * Sn + p]  = r[c * Sn + p] * 0.1f;
            wet[c * Sn + p] = 0.0f;
        }
    }

    float dec = 1.0f;   // DECAY^i
    for (int i = 0; i < 16; ++i) {
        float delay = rd[i] * 3.0f;               // ROOM*12*rsf = 3
        int ys = ((int)floorf(delay)) & 63;
        int xs = ((int)floorf(delay * 2.0f)) & 63;
        float wetw = sigm(rw[i]) * dec * 3.0f;    // *(1+0.4*1.25)*(4*rsf)

        __syncthreads();
        // step 1: r = roll(fb, (ys,xs)) * fades
        for (int kk = 0; kk < 4; ++kk) {
            int p = kk * 1024 + t, y = p >> 6, xx = p & 63;
            int sy = (y - ys) & 63, sx = (xx - xs) & 63;
            float f = fade4(y) * fade4(xx);
            #pragma unroll
            for (int c = 0; c < Cn; ++c)
                r[c * Sn + p] = fb[c * Sn + (sy << 6) + sx] * f;
        }
        __syncthreads();

        // step 2: conv3x3 + damping MLP, results held in registers
        float rf[4][4];
        for (int kk = 0; kk < 4; ++kk) {
            int p = kk * 1024 + t, y = p >> 6, xx = p & 63;
            float rc[4];
            #pragma unroll
            for (int oc = 0; oc < 4; ++oc) rc[oc] = cw[144 + oc];
            for (int dy = -1; dy <= 1; ++dy) {
                int yy = y + dy; if ((unsigned)yy >= 64u) continue;
                for (int dx = -1; dx <= 1; ++dx) {
                    int x2 = xx + dx; if ((unsigned)x2 >= 64u) continue;
                    #pragma unroll
                    for (int ic = 0; ic < 4; ++ic) {
                        float rv = r[ic * Sn + (yy << 6) + x2];
                        #pragma unroll
                        for (int oc = 0; oc < 4; ++oc)
                            rc[oc] += cw[((oc*4 + ic)*3 + (dy+1))*3 + (dx+1)] * rv;
                    }
                }
            }
            float h0 = cw[156], h1 = cw[157];
            #pragma unroll
            for (int ic = 0; ic < 4; ++ic) { h0 += cw[148+ic]*rc[ic]; h1 += cw[152+ic]*rc[ic]; }
            h0 = h0 * sigm(h0);   // silu
            h1 = h1 * sigm(h1);
            #pragma unroll
            for (int oc = 0; oc < 4; ++oc) {
                float dmp = sigm(cw[158 + oc*2]*h0 + cw[159 + oc*2]*h1 + cw[166 + oc]);
                rf[kk][oc] = rc[oc] * dmp;
            }
        }
        __syncthreads();

        // step 3: wet += r*w ; fb += r*0.24   (owned slots only)
        for (int kk = 0; kk < 4; ++kk) {
            int p = kk * 1024 + t;
            #pragma unroll
            for (int c = 0; c < 4; ++c) {
                float vv = rf[kk][c];
                wet[c * Sn + p] += vv * wetw;
                fb[c * Sn + p]  += vv * 0.24f;     // FEEDBACK*1.2*rsf
            }
        }

        // step 4 (i>0): fb += roll(fb, half-delay)*fades * 0.08
        if (i > 0) {
            __syncthreads();
            int ys2 = ((int)floorf(delay * 0.5f)) & 63;
            int xs2 = ((int)floorf(delay)) & 63;
            float pv[4][4];
            for (int kk = 0; kk < 4; ++kk) {
                int p = kk * 1024 + t, y = p >> 6, xx = p & 63;
                int sy = (y - ys2) & 63, sx = (xx - xs2) & 63;
                float f = fade4(y) * fade4(xx);
                #pragma unroll
                for (int c = 0; c < 4; ++c) pv[kk][c] = fb[c * Sn + (sy << 6) + sx] * f;
            }
            __syncthreads();
            for (int kk = 0; kk < 4; ++kk) {
                int p = kk * 1024 + t;
                #pragma unroll
                for (int c = 0; c < 4; ++c) fb[c * Sn + p] += pv[kk][c] * 0.08f;
            }
        }
        dec *= 0.8f;
    }
    __syncthreads();
    for (int kk = 0; kk < 4; ++kk) {
        int p = kk * 1024 + t;
        #pragma unroll
        for (int c = 0; c < 4; ++c)
            wet_out[((b * Cn + c) << 12) + p] = wet[c * Sn + p];
    }
}

// ---------------------------------------------------------------------------
// qkv projection: wet[b,4,s] -> q/k/v laid out [b][head][s]  (hd = 1)
// ---------------------------------------------------------------------------
__global__ void lr_qkv(const float* __restrict__ wet,
                       const float* __restrict__ aiW,   // [12][4]
                       const float* __restrict__ aiB,   // [12]
                       float* __restrict__ q, float* __restrict__ k, float* __restrict__ v) {
    int idx = blockIdx.x * 256 + threadIdx.x;   // 16384
    int b = idx >> 12, s = idx & 4095;
    float w0 = wet[((b*4 + 0) << 12) + s];
    float w1 = wet[((b*4 + 1) << 12) + s];
    float w2 = wet[((b*4 + 2) << 12) + s];
    float w3 = wet[((b*4 + 3) << 12) + s];
    #pragma unroll
    for (int j = 0; j < 12; ++j) {
        float val = aiB[j] + aiW[j*4+0]*w0 + aiW[j*4+1]*w1 + aiW[j*4+2]*w2 + aiW[j*4+3]*w3;
        if (j < 4)      q[((b*4 + j)     << 12) + s] = val;
        else if (j < 8) k[((b*4 + (j-4)) << 12) + s] = val;
        else            v[((b*4 + (j-8)) << 12) + s] = val;
    }
}

// ---------------------------------------------------------------------------
// head_dim=1 softmax attention, WMMA-accumulated.
// grid = (32 qtiles, 16 bh), block = 256 (8 waves, 16 queries/wave).
// K/V staged to LDS with global_load_async_to_lds_b32 (ASYNCcnt path).
// Two independent exp->cvt->WMMA chains hide the WMMA D->A/B hazard window.
// ---------------------------------------------------------------------------
__global__ void lr_attn(const float* __restrict__ q,
                        const float* __restrict__ k,
                        const float* __restrict__ v,
                        float* __restrict__ o) {
    extern __shared__ float sm[];
    float* kbuf  = sm;            // 4096
    float* vbuf  = sm + 4096;     // 4096
    float* stage = sm + 8192;     // 2048 (reduction, then per-wave acc staging)

    const int t     = threadIdx.x;
    const int lane  = t & 31;
    const int wv    = t >> 5;
    const int bh    = blockIdx.y;           // b*4+h
    const int bhoff = bh << 12;

    __builtin_prefetch(&q[bhoff + (blockIdx.x << 7)], 0, 1);   // global_prefetch_b8

    // async LDS fill of K and V (no VGPR round-trip; tracked by ASYNCcnt)
    {
        const float* kp = k + bhoff;
        const float* vp = v + bhoff;
        uint32_t kdst = (uint32_t)(uintptr_t)kbuf;   // LDS byte offset
        uint32_t vdst = (uint32_t)(uintptr_t)vbuf;
        for (int j = t; j < 4096; j += 256) {
            uint32_t goff = (uint32_t)j * 4u;
            uint32_t lk = kdst + goff;
            uint32_t lv = vdst + goff;
            asm volatile("global_load_async_to_lds_b32 %0, %1, %2"
                         :: "v"(lk), "v"(goff), "s"(kp) : "memory");
            asm volatile("global_load_async_to_lds_b32 %0, %1, %2"
                         :: "v"(lv), "v"(goff), "s"(vp) : "memory");
        }
        asm volatile("s_wait_asynccnt 0x0" ::: "memory");
    }
    __syncthreads();

    // block-wide max/min of K (for closed-form safe-softmax row max)
    float lmax = -3.4e38f, lmin = 3.4e38f;
    for (int j = t; j < 4096; j += 256) {
        float kk = kbuf[j];
        lmax = fmaxf(lmax, kk);
        lmin = fminf(lmin, kk);
    }
    stage[t] = lmax; stage[256 + t] = lmin;
    __syncthreads();
    for (int off = 128; off; off >>= 1) {
        if (t < off) {
            stage[t]       = fmaxf(stage[t],       stage[t + off]);
            stage[256 + t] = fminf(stage[256 + t], stage[256 + t + off]);
        }
        __syncthreads();
    }
    const float kmax = stage[0], kmin = stage[256];
    __syncthreads();   // stage gets reused below

    // this wave's 16 queries; row M = lane&15 (A-matrix layout)
    const int qbase = (blockIdx.x << 7) + (wv << 4);
    const float qrow = q[bhoff + qbase + (lane & 15)];
    const float mrow = (qrow >= 0.0f) ? qrow * kmax : qrow * kmin;  // max_j(q*k_j)
    const int   kofs = (lane & 16) ? 8 : 0;    // A K-offset for upper half-wave

    v8f acc0 = {}, acc1 = {};
    #pragma unroll 2
    for (int kc = 0; kc < 128; kc += 2) {
        const int kb0 = kc << 5;
        const int kb1 = kb0 + 32;
        v16h A0, A1;
        #pragma unroll
        for (int e = 0; e < 16; ++e) {
            int K = (e < 8 ? e : e + 8) + kofs;          // 16-bit A 16x32 layout
            A0[e] = (_Float16)__expf(qrow * kbuf[kb0 + K] - mrow);
            A1[e] = (_Float16)__expf(qrow * kbuf[kb1 + K] - mrow);
        }
        v16h B0, B1;
        #pragma unroll
        for (int e = 0; e < 16; ++e) {
            int K = (e < 8 ? e : e + 8) + kofs;          // col 0: v_K, col 1: ones
            B0[e] = ((lane & 15) == 0) ? (_Float16)vbuf[kb0 + K] : (_Float16)(((lane & 15) == 1) ? 1.0f : 0.0f);
            B1[e] = ((lane & 15) == 0) ? (_Float16)vbuf[kb1 + K] : (_Float16)(((lane & 15) == 1) ? 1.0f : 0.0f);
        }
        acc0 = __builtin_amdgcn_wmma_f32_16x16x32_f16(
                   false, A0, false, B0, (short)0, acc0, false, false);
        acc1 = __builtin_amdgcn_wmma_f32_16x16x32_f16(
                   false, A1, false, B1, (short)0, acc1, false, false);
    }

    // stage accumulators, then gather num/den per query (C layout: (M,N) at
    // lane = (M<8 ? N : 16+N), vgpr = M&7)
    #pragma unroll
    for (int j = 0; j < 8; ++j)
        stage[(wv << 8) + (lane << 3) + j] = acc0[j] + acc1[j];
    __syncthreads();
    if (t < 128) {
        int w2 = t >> 4, m = t & 15, base = w2 << 8;
        float num = stage[base + (((m < 8) ? 0 : 16) << 3) + (m & 7)];
        float den = stage[base + (((m < 8) ? 1 : 17) << 3) + (m & 7)];
        o[bhoff + (blockIdx.x << 7) + t] = num / den;
    }
}

// ---------------------------------------------------------------------------
// attention output projection: o[b,head,s] -> wet2[b,c,s]
// ---------------------------------------------------------------------------
__global__ void lr_attnpost(const float* __restrict__ o,
                            const float* __restrict__ aoW,  // [4][4]
                            const float* __restrict__ aoB,
                            float* __restrict__ wet2) {
    int idx = blockIdx.x * 256 + threadIdx.x;   // 16384
    int b = idx >> 12, s = idx & 4095;
    float o0 = o[((b*4+0) << 12) + s];
    float o1 = o[((b*4+1) << 12) + s];
    float o2 = o[((b*4+2) << 12) + s];
    float o3 = o[((b*4+3) << 12) + s];
    #pragma unroll
    for (int c = 0; c < 4; ++c)
        wet2[((b*4+c) << 12) + s] =
            aoB[c] + aoW[c*4+0]*o0 + aoW[c*4+1]*o1 + aoW[c*4+2]*o2 + aoW[c*4+3]*o3;
}

// 3x3 depthwise box blur (zero pad)
__global__ void lr_blur(const float* __restrict__ wet2, float* __restrict__ wet3) {
    int idx = blockIdx.x * 256 + threadIdx.x;   // 65536  [bc][p]
    int p = idx & 4095, bc = idx >> 12;
    int y = p >> 6, xx = p & 63;
    float s = 0.0f;
    for (int dy = -1; dy <= 1; ++dy) {
        int yy = y + dy; if ((unsigned)yy >= 64u) continue;
        for (int dx = -1; dx <= 1; ++dx) {
            int x2 = xx + dx; if ((unsigned)x2 >= 64u) continue;
            s += wet2[(bc << 12) + (yy << 6) + x2];
        }
    }
    wet3[idx] = s * (1.0f / 9.0f);
}

// edge enhance + concat(spatial, wet) 1x1 conv + dry/wet mix
__global__ void lr_final(const float* __restrict__ x,
                         const float* __restrict__ sp,
                         const float* __restrict__ wet3,
                         const float* __restrict__ oW,   // [4][12]
                         const float* __restrict__ oB,
                         float* __restrict__ out) {
    int idx = blockIdx.x * 256 + threadIdx.x;   // 16384
    int b = idx >> 12, p = idx & 4095;
    int y = p >> 6, xx = p & 63;
    float wf[4];
    #pragma unroll
    for (int c = 0; c < 4; ++c) {
        int bc = b*4 + c;
        float center = wet3[(bc << 12) + p];
        float s = 0.0f;
        for (int dy = -1; dy <= 1; ++dy) {
            int yy = y + dy; if ((unsigned)yy >= 64u) continue;
            for (int dx = -1; dx <= 1; ++dx) {
                int x2 = xx + dx; if ((unsigned)x2 >= 64u) continue;
                s += wet3[(bc << 12) + (yy << 6) + x2];
            }
        }
        float edge = 9.0f * center - s;          // 8*c - (neighbors)
        wf[c] = center + edge * 0.04f;           // FEEDBACK*0.2*rsf
    }
    #pragma unroll
    for (int c = 0; c < 4; ++c) {
        float acc = oB[c];
        #pragma unroll
        for (int j = 0; j < 8; ++j) acc += oW[c*12 + j] * sp[((b*8 + j) << 12) + p];
        #pragma unroll
        for (int j = 0; j < 4; ++j) acc += oW[c*12 + 8 + j] * wf[j];
        int oi = ((b*4 + c) << 12) + p;
        out[oi] = x[oi] * 0.7f + acc * 0.3f;     // 1-WET_MIX, WET_MIX
    }
}

extern "C" void kernel_launch(void* const* d_in, const int* in_sizes, int n_in,
                              void* d_out, int out_size, void* d_ws, size_t ws_size,
                              hipStream_t stream) {
    const float* x   = (const float*)d_in[0];
    const float* rw  = (const float*)d_in[1];
    const float* rd  = (const float*)d_in[2];
    // d_in[3] diffusion_strength: unused by the reference
    const float* spW = (const float*)d_in[4];
    const float* spB = (const float*)d_in[5];
    const float* fbW = (const float*)d_in[6];
    const float* fbB = (const float*)d_in[7];
    const float* d1W = (const float*)d_in[8];
    const float* d1B = (const float*)d_in[9];
    const float* d2W = (const float*)d_in[10];
    const float* d2B = (const float*)d_in[11];
    const float* aiW = (const float*)d_in[12];
    const float* aiB = (const float*)d_in[13];
    const float* aoW = (const float*)d_in[14];
    const float* aoB = (const float*)d_in[15];
    const float* oW  = (const float*)d_in[16];
    const float* oB  = (const float*)d_in[17];
    float* out = (float*)d_out;

    float* ws      = (float*)d_ws;
    float* spatial = ws;                    // 131072
    float* wet1    = spatial + 131072;      // 65536
    float* qb      = wet1 + 65536;          // 65536
    float* kb      = qb   + 65536;          // 65536
    float* vb      = kb   + 65536;          // 65536
    float* ob      = vb   + 65536;          // 65536
    float* wet2    = ob   + 65536;          // 65536
    float* wet3    = wet2 + 65536;          // 65536

    lr_spatial<<<512, 256, 0, stream>>>(x, spW, spB, spatial);

    size_t revSm = (size_t)(3 * Cn * Sn + 176) * sizeof(float);   // ~193 KB < 320 KB WGP LDS
    lr_reverb<<<4, 1024, revSm, stream>>>(x, rw, rd, fbW, fbB, d1W, d1B, d2W, d2B, wet1);

    lr_qkv<<<64, 256, 0, stream>>>(wet1, aiW, aiB, qb, kb, vb);

    size_t attSm = (size_t)(4096 + 4096 + 2048) * sizeof(float);  // 40 KB
    lr_attn<<<dim3(32, 16), 256, attSm, stream>>>(qb, kb, vb, ob);

    lr_attnpost<<<64, 256, 0, stream>>>(ob, aoW, aoB, wet2);
    lr_blur<<<256, 256, 0, stream>>>(wet2, wet3);
    lr_final<<<64, 256, 0, stream>>>(x, spatial, wet3, oW, oB, out);
}
